// Linear_88175678587509
// MI455X (gfx1250) — matrix operand order repaired and checked by reading, not compiled, not run on hardware
//
#include <hip/hip_runtime.h>
#include <math.h>

// ---------------------------------------------------------------------------
// Quantized block-sparse linear, MI455X (gfx1250, wave32, WMMA).
//   y = (x @ (w_q*mask)^T) * w_scale + bias ;  fake-quant int8 ; dequant.
// fp16 prepass (96 MiB working set -> L2-resident at 192 MB), then a tiled
// fp16 WMMA GEMM, 128x256 block tile, 64x64 wave tile (4x4 accumulators),
// double-buffered LDS with one barrier per K-step.
// ---------------------------------------------------------------------------

typedef __attribute__((ext_vector_type(16))) _Float16 v16h;
typedef __attribute__((ext_vector_type(8)))  _Float16 v8h;   // 16 bytes
typedef __attribute__((ext_vector_type(4)))  _Float16 v4h;   // 8 bytes
typedef __attribute__((ext_vector_type(8)))  float    v8f;

#define NTOK 8192
#define IDIM 4096
#define ODIM 4096

#define BM   128   // token rows per workgroup tile
#define BO   256   // output cols per workgroup tile
#define BK   32    // K per LDS tile (= one WMMA K step)
#define LDSP 40    // padded LDS row length in halves (BK + 8)

// ---- pack x: f32 -> f16 (4 elements / thread) ------------------------------
__global__ __launch_bounds__(256) void pack_x_kernel(
    const float* __restrict__ x, _Float16* __restrict__ xh) {
  size_t i = (size_t)blockIdx.x * 256u + threadIdx.x;   // in float4 units
  float4 f = ((const float4*)x)[i];
  v4h h;
  h[0] = (_Float16)f.x; h[1] = (_Float16)f.y;
  h[2] = (_Float16)f.z; h[3] = (_Float16)f.w;
  ((v4h*)xh)[i] = h;
}

// ---- pack w: (w_q * block_mask) f32 -> f16. COL_BS=4: one float4 == one
// mask block; ROW_BS=1: mask row == output row. ------------------------------
__global__ __launch_bounds__(256) void pack_w_kernel(
    const float* __restrict__ w_q, const int* __restrict__ mask,
    _Float16* __restrict__ wh) {
  size_t i    = (size_t)blockIdx.x * 256u + threadIdx.x; // float4 units over O*I
  size_t base = i * 4;
  size_t o    = base / IDIM;
  size_t c4   = (base % IDIM) >> 2;
  float  m    = (float)mask[o * (IDIM / 4) + c4];
  float4 f = ((const float4*)w_q)[i];
  v4h h;
  h[0] = (_Float16)(f.x * m); h[1] = (_Float16)(f.y * m);
  h[2] = (_Float16)(f.z * m); h[3] = (_Float16)(f.w * m);
  ((v4h*)wh)[i] = h;
}

// ---- main GEMM: 128x256 tile / 256 threads (8 waves, 2x4 wave grid) --------
// Each wave: 64(M) x 64(O) = 4x4 WMMA accumulators of 16x16 f32.
__global__ __launch_bounds__(256) void gemm_wmma_kernel(
    const _Float16* __restrict__ Xh, const _Float16* __restrict__ Wh,
    const float* __restrict__ bias, const float* __restrict__ w_scale_p,
    const float* __restrict__ out_scale_p, float* __restrict__ out) {
  __shared__ __align__(16) _Float16 As[2][BM * LDSP];  // 2 x 10 KiB
  __shared__ __align__(16) _Float16 Bs[2][BO * LDSP];  // 2 x 20 KiB

  const int tid   = threadIdx.x;
  const int mBase = blockIdx.x * BM;
  const int oBase = blockIdx.y * BO;

  const int lane  = tid & 31;
  const int wave  = tid >> 5;
  const int wm    = (wave >> 2) * 64;  // wave M offset: 0 / 64
  const int wo    = (wave & 3) * 64;   // wave O offset: 0..192
  const int lrow  = lane & 15;         // M (A) / N (B) index within 16
  const int khalf = lane >> 4;         // K-half select per ISA layout

  // Cooperative loads in 16B chunks: A = 512 chunks (2/thr), B = 1024 (4/thr).
  int arow[2], acol[2];
  const v8h* Xg[2];
#pragma unroll
  for (int j = 0; j < 2; ++j) {
    const int c = tid + 256 * j;
    arow[j] = c >> 2; acol[j] = (c & 3) * 8;
    Xg[j] = (const v8h*)(Xh + (size_t)(mBase + arow[j]) * IDIM + acol[j]);
  }
  int brow[4], bcol[4];
  const v8h* Wg[4];
#pragma unroll
  for (int j = 0; j < 4; ++j) {
    const int c = tid + 256 * j;
    brow[j] = c >> 2; bcol[j] = (c & 3) * 8;
    Wg[j] = (const v8h*)(Wh + (size_t)(oBase + brow[j]) * IDIM + bcol[j]);
  }

  v8f acc[4][4] = {};
  v8h xs[2], ws[4];

  // prologue: tile 0 -> buffer 0
#pragma unroll
  for (int j = 0; j < 2; ++j) xs[j] = Xg[j][0];
#pragma unroll
  for (int j = 0; j < 4; ++j) ws[j] = Wg[j][0];
#pragma unroll
  for (int j = 0; j < 2; ++j) *(v8h*)(&As[0][arow[j] * LDSP + acol[j]]) = xs[j];
#pragma unroll
  for (int j = 0; j < 4; ++j) *(v8h*)(&Bs[0][brow[j] * LDSP + bcol[j]]) = ws[j];
  __syncthreads();

  const int NT = IDIM / BK;  // 128 K-tiles
  for (int t = 0; t < NT; ++t) {
    const int cur = t & 1, nxt = cur ^ 1;

    if (t + 1 < NT) {  // prefetch tile t+1 into registers (overlaps compute)
      const int off = (t + 1) * (BK / 8);
#pragma unroll
      for (int j = 0; j < 2; ++j) xs[j] = Xg[j][off];
#pragma unroll
      for (int j = 0; j < 4; ++j) ws[j] = Wg[j][off];
    }

    // B fragments (32x16, 16-bit): lane%16 = N, lane/16 -> K in {0-15,16-31}.
    v16h bf[4];
#pragma unroll
    for (int ni = 0; ni < 4; ++ni) {
      const _Float16* bp = &Bs[cur][(wo + ni * 16 + lrow) * LDSP + khalf * 16];
      v8h lo = *(const v8h*)(bp);
      v8h hi = *(const v8h*)(bp + 8);
      bf[ni] = __builtin_shufflevector(lo, hi, 0, 1, 2, 3, 4, 5, 6, 7,
                                       8, 9, 10, 11, 12, 13, 14, 15);
    }
#pragma unroll
    for (int mi = 0; mi < 4; ++mi) {
      // A fragment (16x32): lane%16 = M, lane/16 -> K in {0-15,16-31}
      // (same contiguous K order as B).
      const _Float16* ap = &As[cur][(wm + mi * 16 + lrow) * LDSP + khalf * 16];
      v8h lo = *(const v8h*)(ap);
      v8h hi = *(const v8h*)(ap + 8);
      v16h af = __builtin_shufflevector(lo, hi, 0, 1, 2, 3, 4, 5, 6, 7,
                                        8, 9, 10, 11, 12, 13, 14, 15);
#pragma unroll
      for (int ni = 0; ni < 4; ++ni) {
        acc[mi][ni] = __builtin_amdgcn_wmma_f32_16x16x32_f16(
            false, af, false, bf[ni], (short)0, acc[mi][ni], false, false);
      }
    }

    if (t + 1 < NT) {  // stage tile t+1 into the other LDS buffer
#pragma unroll
      for (int j = 0; j < 2; ++j)
        *(v8h*)(&As[nxt][arow[j] * LDSP + acol[j]]) = xs[j];
#pragma unroll
      for (int j = 0; j < 4; ++j)
        *(v8h*)(&Bs[nxt][brow[j] * LDSP + bcol[j]]) = ws[j];
    }
    __syncthreads();  // single barrier per K-step
  }

  // ---- epilogue: scale + bias, fake int8 quant, dequant ----
  const float wsc = w_scale_p[0];
  const float osc = out_scale_p[0];
  const float inv_osc = 1.0f / osc;

#pragma unroll
  for (int ni = 0; ni < 4; ++ni) {
    const int ocol = oBase + wo + ni * 16 + lrow;
    const float bv = bias[ocol];
#pragma unroll
    for (int mi = 0; mi < 4; ++mi) {
      const int mrow0 = mBase + wm + mi * 16 + khalf * 8;  // C/D: M=8*half+vgpr
#pragma unroll
      for (int r = 0; r < 8; ++r) {
        float y = acc[mi][ni][r] * wsc + bv;
        float q = rintf(y * inv_osc);
        q = fminf(fmaxf(q, -128.0f), 127.0f);
        out[(size_t)(mrow0 + r) * ODIM + ocol] = q * osc;
      }
    }
  }
}

extern "C" void kernel_launch(void* const* d_in, const int* in_sizes, int n_in,
                              void* d_out, int out_size, void* d_ws, size_t ws_size,
                              hipStream_t stream) {
  const float* x         = (const float*)d_in[0];  // [N, I] f32
  const float* w_q       = (const float*)d_in[1];  // [O, I] f32 (int values)
  const float* bias      = (const float*)d_in[2];  // [O]
  const float* w_scale   = (const float*)d_in[3];  // [1]
  const float* out_scale = (const float*)d_in[4];  // [1]
  const int*   mask      = (const int*)d_in[5];    // [O, I/4]
  float*       out       = (float*)d_out;          // [N, O]

  // workspace: Xh (64 MiB) | Wh (32 MiB)
  _Float16* xh = (_Float16*)d_ws;
  _Float16* wh = xh + (size_t)NTOK * IDIM;

  pack_x_kernel<<<((size_t)NTOK * IDIM / 4) / 256, 256, 0, stream>>>(x, xh);
  pack_w_kernel<<<((size_t)ODIM * IDIM / 4) / 256, 256, 0, stream>>>(w_q, mask, wh);

  dim3 grid(NTOK / BM, ODIM / BO);  // 64 x 16 workgroups
  gemm_wmma_kernel<<<grid, 256, 0, stream>>>(xh, wh, bias, w_scale, out_scale, out);
}
